// SkipgramModel_24335284699350
// MI455X (gfx1250) — matrix operand order repaired and checked by reading, not compiled, not run on hardware
//
#include <hip/hip_runtime.h>
#include <hip/hip_bf16.h>
#include <math.h>

typedef __bf16 bf16;
typedef __attribute__((ext_vector_type(8)))  __bf16 v8bf;
typedef __attribute__((ext_vector_type(16))) __bf16 v16bf;
typedef __attribute__((ext_vector_type(8)))  float  v8f;

#define EMBED 300
#define KPAD  320          // pad K to multiple of 32 (WMMA K) with zeros
#define VOCAB 50000
#define BATCH 4096
#define BM 128
#define BN 128
#define BK 32
#define KTILES (KPAD / BK)                 // 10
#define NTILES_N ((VOCAB + BN - 1) / BN)   // 391
#define NTILES_M (BATCH / BM)              // 32
#define VPAD (NTILES_N * BN)               // 50048: N padded so GEMM loads are unguarded

// ---------------- Kernel 1: gather embedding columns of W1, cast to bf16 ----
// A[b, k] = (k < 300) ? bf16(W1[k, x[b]]) : 0      A is [BATCH, KPAD] row-major
__global__ void gather_cast_A(const float* __restrict__ W1,
                              const int* __restrict__ x,
                              bf16* __restrict__ A) {
  int idx = blockIdx.x * blockDim.x + threadIdx.x;
  if (idx >= BATCH * KPAD) return;
  int b = idx / KPAD;
  int k = idx - b * KPAD;
  float v = 0.0f;
  if (k < EMBED) v = W1[k * VOCAB + x[b]];
  A[idx] = (bf16)v;
}

// ---------------- Kernel 2: cast W2 to bf16, pad K->320 and V->50048 --------
// Bm[v, k] = (v < 50000 && k < 300) ? bf16(W2[v, k]) : 0    Bm is [VPAD, KPAD]
__global__ void cast_W2(const float* __restrict__ W2, bf16* __restrict__ Bm) {
  int idx = blockIdx.x * blockDim.x + threadIdx.x;
  if (idx >= VPAD * KPAD) return;
  int v = idx / KPAD;
  int k = idx - v * KPAD;
  float w = (v < VOCAB && k < EMBED) ? W2[v * EMBED + k] : 0.0f;
  Bm[idx] = (bf16)w;
}

// ---------------- Kernel 3: C = A * Bm^T via v_wmma_f32_16x16x32_bf16 -------
// 256 threads = 8 waves (wave32). Block tile 128x128, K-step 32, LDS double buf.
// Wave (wm in 0..3, wn in 0..1) computes 32(M) x 64(N): 2x4 WMMA tiles.
// B workspace is padded to VPAD rows -> staging loads are fully unconditional
// (pure global_load_b128; no cndmask'd flat path in the hot loop).
__global__ __launch_bounds__(256)
void gemm_bf16(const bf16* __restrict__ A, const bf16* __restrict__ B,
               float* __restrict__ C) {
  // 2 stages * (A 128x32 + B 128x32) bf16 = 32 KB
  __shared__ __align__(16) char smem[2 * (BM * BK + BN * BK) * 2];

  const int tid  = threadIdx.x;
  const int lane = tid & 31;
  const int w    = tid >> 5;
  const int wm   = w & 3;        // wave row (M)
  const int wn   = w >> 2;       // wave col (N), 0..1
  const int half = lane >> 4;    // ISA frag half-select
  const int lr   = lane & 15;

  const int blockN = blockIdx.x;
  const int blockM = blockIdx.y;

  // global->LDS staging: each thread moves 2 float4 of A and 2 of B per stage
  const int rowA0 = tid >> 2;      // 0..63
  const int c8    = tid & 3;       // which 8-bf16 chunk of the 32-wide K slab
  const long aBase = (long)(blockM * BM) * KPAD + c8 * 8;
  const long bBase = (long)(blockN * BN) * KPAD + c8 * 8;

  float4 ar0, ar1, br0, br1;

  auto loadRegs = [&](int kt) {
    const bf16* ap = A + aBase + kt * BK;
    ar0 = *(const float4*)(ap + (long)rowA0 * KPAD);
    ar1 = *(const float4*)(ap + (long)(rowA0 + 64) * KPAD);
    const bf16* bp = B + bBase + kt * BK;
    br0 = *(const float4*)(bp + (long)rowA0 * KPAD);
    br1 = *(const float4*)(bp + (long)(rowA0 + 64) * KPAD);
  };
  auto storeLds = [&](int stage) {
    char* a = smem + stage * 16384;
    char* b = smem + stage * 16384 + 8192;
    *(float4*)(a + rowA0 * 64 + c8 * 16)        = ar0;
    *(float4*)(a + (rowA0 + 64) * 64 + c8 * 16) = ar1;
    *(float4*)(b + rowA0 * 64 + c8 * 16)        = br0;
    *(float4*)(b + (rowA0 + 64) * 64 + c8 * 16) = br1;
  };

  v8f acc[2][4];
  for (int i = 0; i < 2; ++i)
    for (int j = 0; j < 4; ++j)
      for (int r = 0; r < 8; ++r) acc[i][j][r] = 0.0f;

  union Frag { struct { v8bf lo, hi; } h; v16bf v; };

  loadRegs(0);
  storeLds(0);

  for (int kt = 0; kt < KTILES; ++kt) {
    __syncthreads();                         // stage kt&1 visible to all waves
    if (kt + 1 < KTILES) loadRegs(kt + 1);   // overlap vmem with math

    const char* a = smem + (kt & 1) * 16384;
    const char* b = smem + (kt & 1) * 16384 + 8192;

    // A-frag (16x32 bf16, ISA 7.12.2): lane half selects K slab {0..7|8..15}
    // and {16..23|24..31}; each is 8 contiguous bf16 -> one 16B LDS read.
    Frag af[2], bfr[4];
    for (int mi = 0; mi < 2; ++mi) {
      int row = wm * 32 + mi * 16 + lr;
      af[mi].h.lo = *(const v8bf*)(a + row * 64 + half * 16);
      af[mi].h.hi = *(const v8bf*)(a + row * 64 + 32 + half * 16);
    }
    // B-frag (32x16): lanes hold one N column; K = {half*8 + 0..7 | 16 + half*8 + 0..7} (same as A).
    for (int ni = 0; ni < 4; ++ni) {
      int col = wn * 64 + ni * 16 + lr;
      bfr[ni].h.lo = *(const v8bf*)(b + col * 64 + half * 16);
      bfr[ni].h.hi = *(const v8bf*)(b + col * 64 + 32 + half * 16);
    }

    for (int mi = 0; mi < 2; ++mi)
      for (int ni = 0; ni < 4; ++ni)
        acc[mi][ni] = __builtin_amdgcn_wmma_f32_16x16x32_bf16(
            false, af[mi].v, false, bfr[ni].v,
            (short)0, acc[mi][ni], false, false);

    if (kt + 1 < KTILES) storeLds((kt + 1) & 1);  // other buffer; safe pre-sync
  }

  // C layout (ISA 7.12.2): VGPR r, lanes 0-15 -> M=r, lanes 16-31 -> M=r+8
  for (int mi = 0; mi < 2; ++mi) {
    int rowBase = blockM * BM + wm * 32 + mi * 16 + half * 8;
    for (int ni = 0; ni < 4; ++ni) {
      int col = blockN * BN + wn * 64 + ni * 16 + lr;
      if (col < VOCAB) {
        for (int r = 0; r < 8; ++r)
          C[(long)(rowBase + r) * VOCAB + col] = acc[mi][ni][r];
      }
    }
  }
}

// ---------------- Kernel 4: per-row online logsumexp ------------------------
__global__ __launch_bounds__(256)
void row_lse(const float* __restrict__ C, float* __restrict__ lse) {
  __shared__ float sm[256], ss[256];
  const int b = blockIdx.x;
  const float* row = C + (long)b * VOCAB;
  float m = -INFINITY, s = 0.0f;
  for (int v = threadIdx.x; v < VOCAB; v += 256) {
    float xv = row[v];
    if (xv > m) { s = s * __expf(m - xv) + 1.0f; m = xv; }
    else        { s += __expf(xv - m); }
  }
  sm[threadIdx.x] = m; ss[threadIdx.x] = s;
  __syncthreads();
  for (int off = 128; off > 0; off >>= 1) {
    if (threadIdx.x < off) {
      float m1 = sm[threadIdx.x], s1 = ss[threadIdx.x];
      float m2 = sm[threadIdx.x + off], s2 = ss[threadIdx.x + off];
      float M = fmaxf(m1, m2);
      sm[threadIdx.x] = M;
      ss[threadIdx.x] = s1 * __expf(m1 - M) + s2 * __expf(m2 - M);
    }
    __syncthreads();
  }
  if (threadIdx.x == 0) lse[b] = sm[0] + __logf(ss[0]);
}

// ---------------- Kernel 5: out = logits - lse[b] (float4) ------------------
__global__ void sub_lse(float* __restrict__ C, const float* __restrict__ lse) {
  long idx = (long)blockIdx.x * blockDim.x + threadIdx.x;   // float4 index
  const long total = (long)BATCH * (VOCAB / 4);
  if (idx >= total) return;
  int b = (int)(idx / (VOCAB / 4));
  float l = lse[b];
  float4* p = (float4*)C + idx;
  float4 v = *p;
  v.x -= l; v.y -= l; v.z -= l; v.w -= l;
  *p = v;
}

extern "C" void kernel_launch(void* const* d_in, const int* in_sizes, int n_in,
                              void* d_out, int out_size, void* d_ws, size_t ws_size,
                              hipStream_t stream) {
  const int*   x  = (const int*)d_in[0];
  const float* W1 = (const float*)d_in[1];   // [300, 50000]
  const float* W2 = (const float*)d_in[2];   // [50000, 300]
  float* out = (float*)d_out;                // [4096, 50000]
  char*  ws  = (char*)d_ws;

  // workspace layout (256B-aligned offsets)
  bf16*  A   = (bf16*)(ws);                               //  2,621,440 B
  bf16*  Bm  = (bf16*)(ws + 2621440);                     // 32,030,720 B (VPAD*KPAD*2)
  float* lse = (float*)(ws + 2621440 + 32030720);         //     16,384 B

  gather_cast_A<<<(BATCH * KPAD + 255) / 256, 256, 0, stream>>>(W1, x, A);
  cast_W2<<<(VPAD * KPAD + 255) / 256, 256, 0, stream>>>(W2, Bm);

  dim3 grid(NTILES_N, NTILES_M);
  gemm_bf16<<<grid, 256, 0, stream>>>(A, Bm, out);

  row_lse<<<BATCH, 256, 0, stream>>>(out, lse);

  long total4 = (long)BATCH * (VOCAB / 4);
  sub_lse<<<(int)((total4 + 255) / 256), 256, 0, stream>>>(out, lse);
}
